// GastTac_45054206935324
// MI455X (gfx1250) — compile-verified, matrix operand order checked by reading, not run on hardware
//
#include <hip/hip_runtime.h>
#include <hip/hip_bf16.h>

// ---------------------------------------------------------------------------
// GAST-TAC pipeline for MI455X (gfx1250, wave32, WMMA bf16 16x16x32)
//
// All dense math runs through v_wmma_f32_16x16x32_bf16. Activations that feed
// GEMMs are stored bf16 (halves HBM traffic vs f32 on the 268MB tensors);
// accumulation / graph scatter-add / sort keys stay f32. GEMM is double-
// buffered (1 barrier per K-step, global loads overlap WMMA) with
// global_prefetch_b8 two stages ahead.
// ---------------------------------------------------------------------------

typedef unsigned int  u32;
typedef unsigned short u16;

typedef __attribute__((ext_vector_type(16))) __bf16 v16bf;
typedef __attribute__((ext_vector_type(8)))  float  v8f;

union FragAB { v16bf v; u32 u[8]; };

// ---- constants from the reference ----
#define N_NODES 262144
#define N_EDGES 524288
#define IN_F    60
#define IN_FP   64          // padded K for WMMA
#define EMB     256
#define KTOP    96
#define BGR     512         // graphs
#define NPG     512         // nodes per graph
#define C1      128
#define DCATM1  768         // 3*EMB (last channel handled separately)
#define L6      44          // conv6 output length
#define DENSE   11264       // EMB*L6
#define NTAC    50

__device__ __forceinline__ u16 f2bf(float f) {
    u32 u = __float_as_uint(f);
    u32 r = u + 0x7FFFu + ((u >> 16) & 1u);   // round-to-nearest-even
    return (u16)(r >> 16);
}
__device__ __forceinline__ float bf2f(u16 b) {
    return __uint_as_float(((u32)b) << 16);
}

// ===========================================================================
// Generic GEMM: C[M,N](f32) = A[M,K](bf16, row-major) * B[K,N](bf16, row-major)
// BM=64, BN=64*NSUB, BK=32; 256 threads = 8 waves laid out 2(m) x 4(n);
// each wave owns a 32 x (16*NSUB) block of C (2 x NSUB WMMA accumulators).
// Double-buffered LDS stages, one barrier per K-step.
// Requires M%64==0, N%(64*NSUB)==0, K%32==0, K/32 >= 2.
// ===========================================================================
#define BM 64
#define BK 32
#define LDS_K (BK + 2)   // pad to break bank conflicts (68B row stride)

template<int NSUB>
__global__ void __launch_bounds__(256)
k_gemm_bf16(const u16* __restrict__ A, int lda,
            const u16* __restrict__ B, int ldb,
            float* __restrict__ C, int ldc, int K)
{
    constexpr int BNt   = 64 * NSUB;     // block N tile
    constexpr int BREGS = BNt / 16;      // B-tile dwords per thread per stage

    __shared__ u16 sA[2][BM][LDS_K];
    __shared__ u16 sB[2][BNt][LDS_K];

    const int tid  = threadIdx.x;
    const int m0   = blockIdx.x * BM;
    const int n0   = blockIdx.y * BNt;
    const int wave = tid >> 5;
    const int lane = tid & 31;
    const int wm   = (wave & 1) * 32;          // wave M offset within block
    const int wn   = (wave >> 1) * (16 * NSUB);// wave N offset within block
    const int l    = lane & 15;
    const int hi   = lane >> 4;

    v8f zero = {0.f, 0.f, 0.f, 0.f, 0.f, 0.f, 0.f, 0.f};
    v8f acc[2][NSUB];
    #pragma unroll
    for (int ms = 0; ms < 2; ++ms)
        #pragma unroll
        for (int ns = 0; ns < NSUB; ++ns) acc[ms][ns] = zero;

    const int nk = K / BK;

    // ---- prologue: stage tile 0 into buffer 0 ----
    #pragma unroll
    for (int i = 0; i < 4; ++i) {
        int e  = tid + i * 256;                  // dword id over A tile
        int r  = e >> 4;                         // 16 dwords per row
        int kk = (e & 15) << 1;
        u32 val = *(const u32*)(A + (size_t)(m0 + r) * lda + kk);
        *(u32*)(&sA[0][r][kk]) = val;
    }
    #pragma unroll
    for (int i = 0; i < BREGS; ++i) {
        int e  = tid + i * 256;                  // dword id over B tile
        int kk = e / (BNt / 2);
        int nn = (e % (BNt / 2)) << 1;
        u32 val = *(const u32*)(B + (size_t)kk * ldb + n0 + nn);
        sB[0][nn    ][kk] = (u16)(val & 0xFFFFu);
        sB[0][nn + 1][kk] = (u16)(val >> 16);
    }
    __syncthreads();

    u32 aReg[4];
    u32 bReg[BREGS];

    for (int kt = 0; kt < nk; ++kt) {
        const int  cur     = kt & 1;
        const bool hasNext = (kt + 1 < nk);
        const int  k0n     = (kt + 1) * BK;

        // ---- issue global loads for tile kt+1 (fly under the WMMAs) ----
        if (hasNext) {
            #pragma unroll
            for (int i = 0; i < 4; ++i) {
                int e  = tid + i * 256;
                int r  = e >> 4;
                int kk = (e & 15) << 1;
                aReg[i] = *(const u32*)(A + (size_t)(m0 + r) * lda + k0n + kk);
            }
            #pragma unroll
            for (int i = 0; i < BREGS; ++i) {
                int e  = tid + i * 256;
                int kk = e / (BNt / 2);
                int nn = (e % (BNt / 2)) << 1;
                bReg[i] = *(const u32*)(B + (size_t)(k0n + kk) * ldb + n0 + nn);
            }
            // L2 prefetch two stages ahead (global_prefetch_b8)
            if (kt + 2 < nk) {
                int k0p = (kt + 2) * BK;
                __builtin_prefetch(A + (size_t)(m0 + (tid >> 2)) * lda + k0p, 0, 0);
                __builtin_prefetch(B + (size_t)(k0p + (tid >> 3)) * ldb + n0, 0, 0);
            }
        }

        // ---- fragments from stage cur (ISA 7.12.2 16-bit layout) ----
        FragAB af[2], bfr[NSUB];
        #pragma unroll
        for (int ms = 0; ms < 2; ++ms) {
            #pragma unroll
            for (int j = 0; j < 8; ++j) {
                int kk = (j >> 2) * 16 + hi * 8 + (j & 3) * 2;
                af[ms].u[j] = *(const u32*)(&sA[cur][wm + ms * 16 + l][kk]);
            }
        }
        #pragma unroll
        for (int ns = 0; ns < NSUB; ++ns) {
            #pragma unroll
            for (int j = 0; j < 8; ++j) {
                int kk = (j >> 2) * 16 + hi * 8 + (j & 3) * 2;
                bfr[ns].u[j] = *(const u32*)(&sB[cur][wn + ns * 16 + l][kk]);
            }
        }
        #pragma unroll
        for (int ms = 0; ms < 2; ++ms)
            #pragma unroll
            for (int ns = 0; ns < NSUB; ++ns)
                acc[ms][ns] = __builtin_amdgcn_wmma_f32_16x16x32_bf16(
                    false, af[ms].v, false, bfr[ns].v,
                    (short)0, acc[ms][ns], false, false);

        // ---- spill staged registers to the other LDS buffer ----
        if (hasNext) {
            const int nxt = cur ^ 1;
            #pragma unroll
            for (int i = 0; i < 4; ++i) {
                int e  = tid + i * 256;
                int r  = e >> 4;
                int kk = (e & 15) << 1;
                *(u32*)(&sA[nxt][r][kk]) = aReg[i];
            }
            #pragma unroll
            for (int i = 0; i < BREGS; ++i) {
                int e  = tid + i * 256;
                int kk = e / (BNt / 2);
                int nn = (e % (BNt / 2)) << 1;
                sB[nxt][nn    ][kk] = (u16)(bReg[i] & 0xFFFFu);
                sB[nxt][nn + 1][kk] = (u16)(bReg[i] >> 16);
            }
        }
        __syncthreads();
    }

    // ---- store: VGPR r -> M = r + 8*hi, N = lane&15 ----
    #pragma unroll
    for (int ms = 0; ms < 2; ++ms) {
        #pragma unroll
        for (int ns = 0; ns < NSUB; ++ns) {
            int tm = m0 + wm + ms * 16 + hi * 8;
            int tn = n0 + wn + ns * 16 + l;
            #pragma unroll
            for (int r = 0; r < 8; ++r)
                C[(size_t)(tm + r) * ldc + tn] = acc[ms][ns][r];
        }
    }
}

// ===========================================================================
// Conversion / layout kernels
// ===========================================================================
__global__ void __launch_bounds__(256)
k_cast_f2b(const float* __restrict__ s, u16* __restrict__ d, long n)
{
    long i = (long)blockIdx.x * 256 + threadIdx.x;
    if (i < n) d[i] = f2bf(s[i]);
}

// x [N,60] f32 -> xb [N,64] bf16 zero-padded
__global__ void __launch_bounds__(256)
k_cast_x(const float* __restrict__ x, u16* __restrict__ xb)
{
    long i = (long)blockIdx.x * 256 + threadIdx.x;   // over N*64
    int n = (int)(i >> 6), c = (int)(i & 63);
    float v = (c < IN_F) ? x[(size_t)n * IN_F + c] : 0.f;
    xb[i] = f2bf(v);
}

// W1 [60,256] -> [64,256] bf16 zero-padded rows
__global__ void __launch_bounds__(256)
k_cast_w1(const float* __restrict__ w, u16* __restrict__ wb)
{
    int i = blockIdx.x * 256 + threadIdx.x;          // over 64*256
    int r = i >> 8, c = i & 255;
    float v = (r < IN_F) ? w[(size_t)r * EMB + c] : 0.f;
    wb[i] = f2bf(v);
}

// conv5_w [128,769] -> cwt [768,128] bf16 (transposed, drop last channel)
__global__ void __launch_bounds__(256)
k_t_conv5(const float* __restrict__ w, u16* __restrict__ wt)
{
    int i = blockIdx.x * 256 + threadIdx.x;          // over 768*128
    int d = i >> 7, c = i & 127;
    wt[i] = f2bf(w[(size_t)c * 769 + d]);
}
__global__ void __launch_bounds__(128)
k_w768(const float* __restrict__ w, float* __restrict__ wl)
{
    int c = threadIdx.x;
    wl[c] = w[(size_t)c * 769 + 768];
}

// conv6_w [256,128,5] -> w6t [640,256] bf16 : w6t[i*5+t][o] = w[o][i][t]
__global__ void __launch_bounds__(256)
k_t_conv6(const float* __restrict__ w, u16* __restrict__ wt)
{
    int i = blockIdx.x * 256 + threadIdx.x;          // over 640*256
    int f = i >> 8, o = i & 255;
    int ic = f / 5, tau = f % 5;
    wt[(size_t)f * EMB + o] = f2bf(w[((size_t)o * C1 + ic) * 5 + tau]);
}

// ===========================================================================
// GCN graph kernels
// ===========================================================================
__global__ void __launch_bounds__(256)
k_deg(const int* __restrict__ dst, float* __restrict__ deg)
{
    int e = blockIdx.x * 256 + threadIdx.x;
    if (e < N_EDGES) unsafeAtomicAdd(&deg[dst[e]], 1.0f);
}

__global__ void __launch_bounds__(256)
k_dinv(const float* __restrict__ deg, float* __restrict__ dinv)
{
    int n = blockIdx.x * 256 + threadIdx.x;
    dinv[n] = __frsqrt_rn(deg[n] + 1.0f);
}

// one block per edge; 256 channels scatter-added with HW f32 atomics
__global__ void __launch_bounds__(256)
k_agg256(const int* __restrict__ src, const int* __restrict__ dst,
         const float* __restrict__ dinv,
         const float* __restrict__ h, float* __restrict__ agg)
{
    int e = blockIdx.x, c = threadIdx.x;
    int s = src[e], d = dst[e];
    float w = dinv[s] * dinv[d];
    unsafeAtomicAdd(&agg[(size_t)d * EMB + c], w * h[(size_t)s * EMB + c]);
}

__global__ void __launch_bounds__(256)
k_agg1(const int* __restrict__ src, const int* __restrict__ dst,
       const float* __restrict__ dinv,
       const float* __restrict__ h4, float* __restrict__ agg4)
{
    int e = blockIdx.x * 256 + threadIdx.x;
    if (e < N_EDGES) {
        int s = src[e], d = dst[e];
        unsafeAtomicAdd(&agg4[d], dinv[s] * dinv[d] * h4[s]);
    }
}

// out_b = bf16(tanh(agg + dinv^2*h + b))  (next-layer GEMM operand)
__global__ void __launch_bounds__(256)
k_finalize256(const float* __restrict__ h, const float* __restrict__ agg,
              const float* __restrict__ dinv, const float* __restrict__ b,
              u16* __restrict__ out_b)
{
    long i = (long)blockIdx.x * 256 + threadIdx.x;   // over N*256
    int n = (int)(i >> 8), c = (int)(i & 255);
    float di = dinv[n];
    float v = tanhf(agg[i] + di * di * h[i] + b[c]);
    out_b[i] = f2bf(v);
}

__global__ void __launch_bounds__(256)
k_finalize1(const float* __restrict__ h4, const float* __restrict__ agg4,
            const float* __restrict__ dinv, const float* __restrict__ b4,
            float* __restrict__ x4)
{
    int n = blockIdx.x * 256 + threadIdx.x;
    float di = dinv[n];
    x4[n] = tanhf(agg4[n] + di * di * h4[n] + b4[0]);
}

// layer-4 projection (EMB -> 1): one node per thread
__global__ void __launch_bounds__(256)
k_dot1(const u16* __restrict__ x3b, const float* __restrict__ W4,
       float* __restrict__ h4)
{
    int n = blockIdx.x * 256 + threadIdx.x;
    const u16* row = x3b + (size_t)n * EMB;
    float s = 0.f;
    #pragma unroll 8
    for (int k = 0; k < EMB; ++k) s += bf2f(row[k]) * W4[k];
    h4[n] = s;
}

// ===========================================================================
// Per-graph top-K sort (descending by x4). One block (256 thr) per graph,
// 512-element bitonic sort in LDS.
// ===========================================================================
__global__ void __launch_bounds__(256)
k_sort_topk(const float* __restrict__ x4, int* __restrict__ topk)
{
    __shared__ float key[NPG];
    __shared__ int   idx[NPG];
    int g = blockIdx.x, tid = threadIdx.x;
    for (int i = tid; i < NPG; i += 256) {
        key[i] = x4[g * NPG + i];
        idx[i] = g * NPG + i;
    }
    __syncthreads();
    for (int k = 2; k <= NPG; k <<= 1) {
        for (int j = k >> 1; j > 0; j >>= 1) {
            for (int t = tid; t < NPG; t += 256) {
                int ixj = t ^ j;
                if (ixj > t) {
                    bool up = ((t & k) == 0);          // descending overall
                    bool sw = up ? (key[t] < key[ixj]) : (key[t] > key[ixj]);
                    if (sw) {
                        float fk = key[t]; key[t] = key[ixj]; key[ixj] = fk;
                        int   ii = idx[t]; idx[t] = idx[ixj]; idx[ixj] = ii;
                    }
                }
            }
            __syncthreads();
        }
    }
    for (int t = tid; t < KTOP; t += 256) topk[g * KTOP + t] = idx[t];
}

// gather pooled rows: pooledb[row, 0:768] = [x1b|x2b|x3b][node], plast = x4
__global__ void __launch_bounds__(256)
k_gather(const int* __restrict__ topk,
         const u16* __restrict__ x1b, const u16* __restrict__ x2b,
         const u16* __restrict__ x3b, const float* __restrict__ x4,
         u16* __restrict__ pooledb, float* __restrict__ plast)
{
    int row = blockIdx.x;           // 0 .. B*K-1
    int c   = threadIdx.x;
    int node = topk[row];
    size_t o = (size_t)row * DCATM1;
    pooledb[o + c]        = x1b[(size_t)node * EMB + c];
    pooledb[o + 256 + c]  = x2b[(size_t)node * EMB + c];
    pooledb[o + 512 + c]  = x3b[(size_t)node * EMB + c];
    if (c == 0) plast[row] = x4[node];
}

// conv5 epilogue: add last-channel rank-1 term + bias, ReLU, MaxPool1d(2)
// C5 [B*K,128] -> pool48 [B,128,48]
__global__ void __launch_bounds__(256)
k_conv5_ep(const float* __restrict__ C5, const float* __restrict__ plast,
           const float* __restrict__ w768, const float* __restrict__ b5,
           float* __restrict__ pool48)
{
    long i = (long)blockIdx.x * 256 + threadIdx.x;   // over B*128*48
    int j = (int)(i % 48);
    int c = (int)((i / 48) % C1);
    int g = (int)(i / (48 * C1));
    int r0 = g * KTOP + 2 * j, r1 = r0 + 1;
    float a0 = C5[(size_t)r0 * C1 + c] + plast[r0] * w768[c] + b5[c];
    float a1 = C5[(size_t)r1 * C1 + c] + plast[r1] * w768[c] + b5[c];
    a0 = fmaxf(a0, 0.f); a1 = fmaxf(a1, 0.f);
    pool48[((size_t)g * C1 + c) * 48 + j] = fmaxf(a0, a1);
}

// im2col for conv6: A6[g*44+t, i*5+tau] = bf16(pool48[g, i, t+tau])
__global__ void __launch_bounds__(256)
k_im2col(const float* __restrict__ pool48, u16* __restrict__ A6)
{
    long i = (long)blockIdx.x * 256 + threadIdx.x;   // over (B*44)*640
    int f   = (int)(i % 640);
    int row = (int)(i / 640);
    int g = row / L6, t = row % L6;
    int ic = f / 5, tau = f % 5;
    A6[i] = f2bf(pool48[((size_t)g * C1 + ic) * 48 + t + tau]);
}

// conv6 epilogue: ReLU(+bias) and permuted flatten into fc1 operand (bf16)
// C6 [(g*44+t), o] -> h1b[g, o*44+t]
__global__ void __launch_bounds__(256)
k_conv6_ep(const float* __restrict__ C6, const float* __restrict__ b6,
           u16* __restrict__ h1b)
{
    long i = (long)blockIdx.x * 256 + threadIdx.x;   // over B*256*44
    int t = (int)(i % L6);
    int c = (int)((i / L6) % EMB);
    int g = (int)(i / (L6 * EMB));
    float v = fmaxf(C6[((size_t)g * L6 + t) * EMB + c] + b6[c], 0.f);
    h1b[(size_t)g * DENSE + c * L6 + t] = f2bf(v);
}

__global__ void __launch_bounds__(256)
k_fc1_ep(const float* __restrict__ Cf, const float* __restrict__ b,
         float* __restrict__ h2)
{
    int i = blockIdx.x * 256 + threadIdx.x;          // over B*256
    h2[i] = fmaxf(Cf[i] + b[i & 255], 0.f);
}

__global__ void __launch_bounds__(256)
k_fc2(const float* __restrict__ h2, const float* __restrict__ w,
      const float* __restrict__ b, float* __restrict__ out)
{
    int i = blockIdx.x * 256 + threadIdx.x;          // over B*64
    int g = i >> 6, t = i & 63;
    if (t < NTAC) {
        float s = b[t];
        #pragma unroll 8
        for (int c = 0; c < EMB; ++c) s += h2[(size_t)g * EMB + c] * w[(size_t)c * NTAC + t];
        out[(size_t)g * NTAC + t] = s;
    }
}

// ===========================================================================
// Host-side orchestration
// ===========================================================================
extern "C" void kernel_launch(void* const* d_in, const int* in_sizes, int n_in,
                              void* d_out, int out_size, void* d_ws, size_t ws_size,
                              hipStream_t stream) {
    (void)in_sizes; (void)n_in; (void)out_size; (void)ws_size;

    const float* x       = (const float*)d_in[0];
    const int*   ei      = (const int*)  d_in[1];
    const float* W1      = (const float*)d_in[2];
    const float* b1      = (const float*)d_in[3];
    const float* W2      = (const float*)d_in[4];
    const float* b2      = (const float*)d_in[5];
    const float* W3      = (const float*)d_in[6];
    const float* b3      = (const float*)d_in[7];
    const float* W4      = (const float*)d_in[8];
    const float* b4      = (const float*)d_in[9];
    const float* conv5_w = (const float*)d_in[10];
    const float* conv5_b = (const float*)d_in[11];
    const float* conv6_w = (const float*)d_in[12];
    const float* conv6_b = (const float*)d_in[13];
    const float* fc1_w   = (const float*)d_in[14];
    const float* fc1_b   = (const float*)d_in[15];
    const float* fc2_w   = (const float*)d_in[16];
    const float* fc2_b   = (const float*)d_in[17];
    float* out = (float*)d_out;

    const int* src = ei;
    const int* dst = ei + N_EDGES;

    // ---- workspace bump allocator ----
    char* base = (char*)d_ws;
    size_t off = 0;
    auto alloc = [&](size_t bytes) -> void* {
        void* p = base + off;
        off += (bytes + 255) & ~(size_t)255;
        return p;
    };
    const int BK_ROWS = BGR * KTOP;   // 49152
    u16*   xb    = (u16*)  alloc((size_t)N_NODES * IN_FP * 2);
    u16*   W1b   = (u16*)  alloc((size_t)IN_FP * EMB * 2);
    u16*   W2b   = (u16*)  alloc((size_t)EMB * EMB * 2);
    u16*   W3b   = (u16*)  alloc((size_t)EMB * EMB * 2);
    u16*   x1b   = (u16*)  alloc((size_t)N_NODES * EMB * 2);
    u16*   x2b   = (u16*)  alloc((size_t)N_NODES * EMB * 2);
    u16*   x3b   = (u16*)  alloc((size_t)N_NODES * EMB * 2);
    float* h     = (float*)alloc((size_t)N_NODES * EMB * 4);
    float* agg   = (float*)alloc((size_t)N_NODES * EMB * 4);
    float* deg   = (float*)alloc((size_t)N_NODES * 4);
    float* dinv  = (float*)alloc((size_t)N_NODES * 4);
    float* h4    = (float*)alloc((size_t)N_NODES * 4);
    float* agg4  = (float*)alloc((size_t)N_NODES * 4);
    float* x4    = (float*)alloc((size_t)N_NODES * 4);
    int*   topk  = (int*)  alloc((size_t)BGR * KTOP * 4);
    u16*   pooledb = (u16*)alloc((size_t)BK_ROWS * DCATM1 * 2);
    float* plast = (float*)alloc((size_t)BK_ROWS * 4);
    u16*   cwt   = (u16*)  alloc((size_t)DCATM1 * C1 * 2);
    float* w768  = (float*)alloc((size_t)C1 * 4);
    float* C5    = (float*)alloc((size_t)BK_ROWS * C1 * 4);
    float* pool48= (float*)alloc((size_t)BGR * C1 * 48 * 4);
    u16*   A6    = (u16*)  alloc((size_t)BGR * L6 * 640 * 2);
    u16*   w6t   = (u16*)  alloc((size_t)640 * EMB * 2);
    float* C6    = (float*)alloc((size_t)BGR * L6 * EMB * 4);
    u16*   h1b   = (u16*)  alloc((size_t)BGR * DENSE * 2);
    u16*   fc1wb = (u16*)  alloc((size_t)DENSE * EMB * 2);
    float* Cf1   = (float*)alloc((size_t)BGR * EMB * 4);
    float* h2    = (float*)alloc((size_t)BGR * EMB * 4);

    // ---- weight / input conversion ----
    k_cast_x  <<<(N_NODES * IN_FP) / 256, 256, 0, stream>>>(x, xb);
    k_cast_w1 <<<(IN_FP * EMB) / 256, 256, 0, stream>>>(W1, W1b);
    k_cast_f2b<<<(EMB * EMB) / 256, 256, 0, stream>>>(W2, W2b, (long)EMB * EMB);
    k_cast_f2b<<<(EMB * EMB) / 256, 256, 0, stream>>>(W3, W3b, (long)EMB * EMB);
    k_t_conv5 <<<(DCATM1 * C1) / 256, 256, 0, stream>>>(conv5_w, cwt);
    k_w768    <<<1, 128, 0, stream>>>(conv5_w, w768);
    k_t_conv6 <<<(640 * EMB) / 256, 256, 0, stream>>>(conv6_w, w6t);
    k_cast_f2b<<<((long)DENSE * EMB) / 256, 256, 0, stream>>>(fc1_w, fc1wb, (long)DENSE * EMB);

    // ---- degree normalization ----
    hipMemsetAsync(deg, 0, (size_t)N_NODES * 4, stream);
    k_deg <<<N_EDGES / 256, 256, 0, stream>>>(dst, deg);
    k_dinv<<<N_NODES / 256, 256, 0, stream>>>(deg, dinv);

    dim3 gemm_blk(256);
    const long NE = (long)N_NODES * EMB;

    // ---- GCN layer 1 (K=64 padded), N=256 -> NSUB=4, grid.y=1 ----
    k_gemm_bf16<4><<<dim3(N_NODES / BM, 1), gemm_blk, 0, stream>>>(xb, IN_FP, W1b, EMB, h, EMB, IN_FP);
    hipMemsetAsync(agg, 0, NE * 4, stream);
    k_agg256     <<<N_EDGES, 256, 0, stream>>>(src, dst, dinv, h, agg);
    k_finalize256<<<NE / 256, 256, 0, stream>>>(h, agg, dinv, b1, x1b);

    // ---- GCN layer 2 ----
    k_gemm_bf16<4><<<dim3(N_NODES / BM, 1), gemm_blk, 0, stream>>>(x1b, EMB, W2b, EMB, h, EMB, EMB);
    hipMemsetAsync(agg, 0, NE * 4, stream);
    k_agg256     <<<N_EDGES, 256, 0, stream>>>(src, dst, dinv, h, agg);
    k_finalize256<<<NE / 256, 256, 0, stream>>>(h, agg, dinv, b2, x2b);

    // ---- GCN layer 3 ----
    k_gemm_bf16<4><<<dim3(N_NODES / BM, 1), gemm_blk, 0, stream>>>(x2b, EMB, W3b, EMB, h, EMB, EMB);
    hipMemsetAsync(agg, 0, NE * 4, stream);
    k_agg256     <<<N_EDGES, 256, 0, stream>>>(src, dst, dinv, h, agg);
    k_finalize256<<<NE / 256, 256, 0, stream>>>(h, agg, dinv, b3, x3b);

    // ---- GCN layer 4 (EMB -> 1) ----
    k_dot1<<<N_NODES / 256, 256, 0, stream>>>(x3b, W4, h4);
    hipMemsetAsync(agg4, 0, (size_t)N_NODES * 4, stream);
    k_agg1     <<<N_EDGES / 256, 256, 0, stream>>>(src, dst, dinv, h4, agg4);
    k_finalize1<<<N_NODES / 256, 256, 0, stream>>>(h4, agg4, dinv, b4, x4);

    // ---- sort-pool ----
    k_sort_topk<<<BGR, 256, 0, stream>>>(x4, topk);
    k_gather   <<<BK_ROWS, 256, 0, stream>>>(topk, x1b, x2b, x3b, x4, pooledb, plast);

    // ---- conv5 as GEMM [49152,768] x [768,128] (N=128 -> NSUB=2) ----
    k_gemm_bf16<2><<<dim3(BK_ROWS / BM, 1), gemm_blk, 0, stream>>>(pooledb, DCATM1, cwt, C1, C5, C1, DCATM1);
    k_conv5_ep <<<((long)BGR * C1 * 48) / 256, 256, 0, stream>>>(C5, plast, w768, conv5_b, pool48);

    // ---- conv6 as im2col GEMM [22528,640] x [640,256] ----
    k_im2col      <<<((long)BGR * L6 * 640) / 256, 256, 0, stream>>>(pool48, A6);
    k_gemm_bf16<4><<<dim3((BGR * L6) / BM, 1), gemm_blk, 0, stream>>>(A6, 640, w6t, EMB, C6, EMB, 640);
    k_conv6_ep    <<<((long)BGR * EMB * L6) / 256, 256, 0, stream>>>(C6, conv6_b, h1b);

    // ---- fc1 GEMM [512,11264] x [11264,256] ----
    k_gemm_bf16<4><<<dim3(BGR / BM, 1), gemm_blk, 0, stream>>>(h1b, DENSE, fc1wb, EMB, Cf1, EMB, DENSE);
    k_fc1_ep      <<<(BGR * EMB) / 256, 256, 0, stream>>>(Cf1, fc1_b, h2);

    // ---- fc2 -> logits [512,50] ----
    k_fc2<<<(BGR * 64) / 256, 256, 0, stream>>>(h2, fc2_w, fc2_b, out);
}
